// SARAEngine_4389456577451
// MI455X (gfx1250) — matrix operand order repaired and checked by reading, not compiled, not run on hardware
//
#include <hip/hip_runtime.h>
#include <hip/hip_bf16.h>
#include <cstdint>

typedef __attribute__((ext_vector_type(16))) _Float16 v16h;
typedef __attribute__((ext_vector_type(8)))  float    v8f;

#define WMMA_F16(a, b, c) \
  __builtin_amdgcn_wmma_f32_16x16x32_f16(false, (a), false, (b), (short)0, (c), false, false)

__device__ __forceinline__ float sigm_(float x) { return 1.0f / (1.0f + __expf(-x)); }

// -------- CDNA5 async global->LDS copy (ASYNCcnt) -------------------------
__device__ __forceinline__ uint32_t lds_off32(const void* p) {
  // flat LDS aperture keeps the byte offset in addr[31:0]
  return (uint32_t)(uintptr_t)p;
}
__device__ __forceinline__ void async_copy_b128(const void* gsrc, uint32_t lds_dst) {
  asm volatile("global_load_async_to_lds_b128 %0, %1, off"
               :: "v"(lds_dst), "v"((uint64_t)(uintptr_t)gsrc)
               : "memory");
}
template <int N>
__device__ __forceinline__ void wait_async() {
#if __has_builtin(__builtin_amdgcn_s_wait_asynccnt)
  __builtin_amdgcn_s_wait_asynccnt(N);
#else
  asm volatile("s_wait_asynccnt %0" :: "n"(N));
#endif
}

// ---------------------------------------------------------------------------
// Split f32 -> (hi, lo) f16 pair (all weight matrices, done once per call)
// ---------------------------------------------------------------------------
__global__ __launch_bounds__(256) void split_kernel(const float* __restrict__ W,
                                                    _Float16* __restrict__ hi,
                                                    _Float16* __restrict__ lo, int n) {
  int i = blockIdx.x * 256 + threadIdx.x;
  if (i < n) {
    float x = W[i];
    _Float16 h = (_Float16)x;
    hi[i] = h;
    lo[i] = (_Float16)(x - (float)h);
  }
}

// ---------------------------------------------------------------------------
// Split-precision WMMA GEMM:  C[M,N] = A[M,K] @ W[N,K]^T + bias[N]
// Block: 256 threads = 8 waves; tile 128(M) x 64(N); K stepped by 32.
// B (pre-split f16 hi/lo) is staged into double-buffered LDS with
// global_load_async_to_lds_b128, pipelined one K-step ahead; last K-step is
// peeled so the steady-state loop has no branches around the async waits.
// A_IS_F16: A is exact f16 (spikes) -> 2 WMMAs/tile, else f32 split -> 3.
// ---------------------------------------------------------------------------
template <bool A_IS_F16>
__global__ __launch_bounds__(256) void gemm_wmma_kernel(
    const void* __restrict__ Aptr,
    const _Float16* __restrict__ Whi, const _Float16* __restrict__ Wlo,
    const float* __restrict__ bias, float* __restrict__ C, int M, int N, int K) {
  __shared__ _Float16 sBhi[2][64 * 32];
  __shared__ _Float16 sBlo[2][64 * 32];

  const int tid  = threadIdx.x;
  const int wave = tid >> 5;
  const int lane = tid & 31;
  const int mBase = blockIdx.y * 128 + wave * 16;
  const int nBase = blockIdx.x * 64;

  const int lm     = lane & 15;
  const int hiHalf = lane >> 4;
  const int kA     = hiHalf ? 8 : 0;   // A fragment k sub-offset per lane half
  const int kB     = hiHalf ? 8 : 0;   // B fragment k sub-offset per lane half

  v8f acc[4] = {};

  // async-stage indices: thread -> (n = tid>>2 in 0..63, 8 consecutive k = 16B)
  const int ln = tid >> 2;
  const int lk = (tid & 3) * 8;
  const int gn = nBase + ln;
  const bool nValid = (gn < N);
  const _Float16* gBhi = Whi + (size_t)gn * K + lk;
  const _Float16* gBlo = Wlo + (size_t)gn * K + lk;
  const uint32_t ldsHi[2] = {lds_off32(&sBhi[0][ln * 32 + lk]),
                             lds_off32(&sBhi[1][ln * 32 + lk])};
  const uint32_t ldsLo[2] = {lds_off32(&sBlo[0][ln * 32 + lk]),
                             lds_off32(&sBlo[1][ln * 32 + lk])};

  // consume one staged 32-wide K slice from LDS buffer `buf`
  auto consume = [&](int k0, int buf) {
    // ---- A fragment (16x32, per ISA layout) ----
    v16h ah, al;
    if (A_IS_F16) {
      const _Float16* pa =
          (const _Float16*)Aptr + (size_t)(mBase + lm) * K + k0 + kA;
#pragma unroll
      for (int v = 0; v < 4; ++v) {
        ah[2 * v]     = pa[2 * v];
        ah[2 * v + 1] = pa[2 * v + 1];
        ah[8 + 2 * v] = pa[16 + 2 * v];
        ah[9 + 2 * v] = pa[16 + 2 * v + 1];
      }
    } else {
      const float* pa = (const float*)Aptr + (size_t)(mBase + lm) * K + k0 + kA;
#pragma unroll
      for (int v = 0; v < 8; ++v) {
        int src = (v < 4) ? 2 * v : 16 + 2 * (v - 4);
        float x0 = pa[src], x1 = pa[src + 1];
        _Float16 h0 = (_Float16)x0, h1 = (_Float16)x1;
        ah[2 * v] = h0;
        ah[2 * v + 1] = h1;
        al[2 * v] = (_Float16)(x0 - (float)h0);
        al[2 * v + 1] = (_Float16)(x1 - (float)h1);
      }
    }

    // ---- 4 N-tiles: build B frags from LDS, issue WMMAs ----
#pragma unroll
    for (int nt = 0; nt < 4; ++nt) {
      const _Float16* ph = &sBhi[buf][(nt * 16 + lm) * 32 + kB];
      const _Float16* pl = &sBlo[buf][(nt * 16 + lm) * 32 + kB];
      v16h bh, bl;
#pragma unroll
      for (int v = 0; v < 8; ++v) {
        bh[v] = ph[v]; bh[8 + v] = ph[16 + v];
        bl[v] = pl[v]; bl[8 + v] = pl[16 + v];
      }
      acc[nt] = WMMA_F16(ah, bh, acc[nt]);
      acc[nt] = WMMA_F16(ah, bl, acc[nt]);
      if (!A_IS_F16) acc[nt] = WMMA_F16(al, bh, acc[nt]);
    }
  };

  // prologue: stage k0 = 0 into buffer 0
  if (nValid) {
    async_copy_b128(gBhi, ldsHi[0]);
    async_copy_b128(gBlo, ldsLo[0]);
  }

  // steady state: always issue next slice, wait <=2 (current group complete)
  int buf = 0;
  for (int k0 = 0; k0 + 32 < K; k0 += 32, buf ^= 1) {
    if (nValid) {
      async_copy_b128(gBhi + k0 + 32, ldsHi[buf ^ 1]);
      async_copy_b128(gBlo + k0 + 32, ldsLo[buf ^ 1]);
    }
    wait_async<2>();
    __syncthreads();
    consume(k0, buf);
    __syncthreads();
  }
  // peeled last slice
  wait_async<0>();
  __syncthreads();
  consume(K - 32, buf);

  // ---- epilogue: +bias, store (C/D layout: VGPR i -> M=i(+8), N=lane&15) ----
  const int mrow = mBase + (hiHalf ? 8 : 0);
#pragma unroll
  for (int nt = 0; nt < 4; ++nt) {
    int n = nBase + nt * 16 + lm;
    if (n < N) {
      float bv = bias[n];
#pragma unroll
      for (int i = 0; i < 8; ++i) C[(size_t)(mrow + i) * N + n] = acc[nt][i] + bv;
    }
  }
}

// ---------------------------------------------------------------------------
// LayerNorm: one block (256 thr) per row
// ---------------------------------------------------------------------------
__global__ __launch_bounds__(256) void layernorm_kernel(const float* __restrict__ x,
                                                        const float* __restrict__ g,
                                                        const float* __restrict__ b,
                                                        float* __restrict__ y, int D) {
  __shared__ float red[256];
  const int row = blockIdx.x;
  const int tid = threadIdx.x;
  const float* xr = x + (size_t)row * D;

  float s = 0.f;
  for (int i = tid; i < D; i += 256) s += xr[i];
  red[tid] = s;
  __syncthreads();
  for (int off = 128; off > 0; off >>= 1) {
    if (tid < off) red[tid] += red[tid + off];
    __syncthreads();
  }
  const float mu = red[0] / (float)D;
  __syncthreads();

  float vs = 0.f;
  for (int i = tid; i < D; i += 256) { float d = xr[i] - mu; vs += d * d; }
  red[tid] = vs;
  __syncthreads();
  for (int off = 128; off > 0; off >>= 1) {
    if (tid < off) red[tid] += red[tid + off];
    __syncthreads();
  }
  const float rstd = rsqrtf(red[0] / (float)D + 1e-5f);
  __syncthreads();

  for (int i = tid; i < D; i += 256) y[(size_t)row * D + i] = (xr[i] - mu) * rstd * g[i] + b[i];
}

// ---------------------------------------------------------------------------
// LIF: 20-step leaky integrate & fire, emit f16 spikes + per-block spike count
// ---------------------------------------------------------------------------
__global__ __launch_bounds__(256) void lif_kernel(const float* __restrict__ cur,
                                                  _Float16* __restrict__ spikes,
                                                  float* __restrict__ partial) {
  __shared__ float red[256];
  const int idx = blockIdx.x * 256 + threadIdx.x;  // b*1024 + j
  const int b = idx >> 10;
  const int j = idx & 1023;
  const float c = cur[idx];
  float v = 0.f, cnt = 0.f;
#pragma unroll
  for (int t = 0; t < 20; ++t) {
    v += (c - v) * 0.5f;  // TAU = 2
    float s = (v - 1.0f > 0.f) ? 1.f : 0.f;
    v -= s;  // VTH = 1
    spikes[((size_t)(b * 20 + t) << 10) + j] = (_Float16)s;
    cnt += s;
  }
  red[threadIdx.x] = cnt;
  __syncthreads();
  for (int off = 128; off > 0; off >>= 1) {
    if (threadIdx.x < off) red[threadIdx.x] += red[threadIdx.x + off];
    __syncthreads();
  }
  if (threadIdx.x == 0) partial[blockIdx.x] = red[0];
}

// ---------------------------------------------------------------------------
// Fused SSM: causal depthwise conv(4)+SiLU + dt GEMM(K=64)+softplus +
// 20-step/16-state selective scan + final gating. One thread per (b, channel).
// ---------------------------------------------------------------------------
__global__ __launch_bounds__(256) void ssm_kernel(
    const float* __restrict__ proj, const float* __restrict__ conv_w,
    const float* __restrict__ conv_b, const float* __restrict__ dt_W,
    const float* __restrict__ dt_b, const float* __restrict__ A_log,
    const float* __restrict__ Dv, float* __restrict__ fm) {
  __shared__ float sdtr[20][64];
  __shared__ float sB[20][16];
  __shared__ float sC[16];
  const int b = blockIdx.y;
  const int tid = threadIdx.x;
  const int c = blockIdx.x * 256 + tid;
  const size_t rowBase = (size_t)b * 20;

  for (int i = tid; i < 20 * 64; i += 256) {
    int t = i >> 6, r = i & 63;
    sdtr[t][r] = proj[(rowBase + t) * 4192 + 4128 + r];
  }
  for (int i = tid; i < 20 * 16; i += 256) {
    int t = i >> 4, s = i & 15;
    sB[t][s] = proj[(rowBase + t) * 4192 + 4096 + s];
  }
  if (tid < 16) sC[tid] = proj[(rowBase + 19) * 4192 + 4112 + tid];
  __syncthreads();

  const float w0 = conv_w[c * 4 + 0], w1 = conv_w[c * 4 + 1];
  const float w2 = conv_w[c * 4 + 2], w3 = conv_w[c * 4 + 3];
  const float cb = conv_b[c];
  const float dtb = dt_b[c];
  float dtw[64];
#pragma unroll
  for (int r = 0; r < 64; ++r) dtw[r] = dt_W[c * 64 + r];
  float aneg[16];
#pragma unroll
  for (int s = 0; s < 16; ++s) aneg[s] = -__expf(A_log[c * 16 + s]);
  float h[16];
#pragma unroll
  for (int s = 0; s < 16; ++s) h[s] = 0.f;

  float x1 = 0.f, x2 = 0.f, x3 = 0.f;
  float xcLast = 0.f;
  for (int t = 0; t < 20; ++t) {
    float x0 = proj[(rowBase + t) * 4192 + 2048 + c];
    float conv = w0 * x3 + w1 * x2 + w2 * x1 + w3 * x0 + cb;  // causal, pad 3
    float xc = conv * sigm_(conv);                            // SiLU
    float dacc = dtb;
#pragma unroll
    for (int r = 0; r < 64; ++r) dacc += sdtr[t][r] * dtw[r];
    float dt = (dacc > 20.f) ? dacc : log1pf(__expf(dacc));   // softplus
#pragma unroll
    for (int s = 0; s < 16; ++s)
      h[s] = __expf(aneg[s] * dt) * h[s] + (dt * sB[t][s]) * xc;
    x3 = x2; x2 = x1; x1 = x0;
    xcLast = xc;
  }
  float y = 0.f;
#pragma unroll
  for (int s = 0; s < 16; ++s) y += h[s] * sC[s];

  float z = proj[(rowBase + 19) * 4192 + c];
  fm[(size_t)b * 2048 + c] = (y + xcLast * Dv[c]) * (z * sigm_(z));
}

// ---------------------------------------------------------------------------
// GRU gate update. ghRowStride==0 => gh is broadcast b_hh (h==0 first iter).
// ---------------------------------------------------------------------------
__global__ __launch_bounds__(256) void gru_gate_kernel(const float* __restrict__ gi,
                                                       const float* __restrict__ gh,
                                                       int ghRowStride,
                                                       float* __restrict__ h, int first) {
  int idx = blockIdx.x * 256 + threadIdx.x;  // b*1024 + j
  int b = idx >> 10, j = idx & 1023;
  size_t gb = (size_t)b * 3072;
  float ir = gi[gb + j], iz = gi[gb + 1024 + j], in = gi[gb + 2048 + j];
  size_t hb = (size_t)b * ghRowStride;
  float hr = gh[hb + j], hz = gh[hb + 1024 + j], hn = gh[hb + 2048 + j];
  float hp = first ? 0.f : h[idx];
  float r = sigm_(ir + hr);
  float z = sigm_(iz + hz);
  float n = tanhf(in + r * hn);
  h[idx] = (1.f - z) * n + z * hp;
}

// ---------------------------------------------------------------------------
// Reduce LIF partials -> rate; write trailing scalars of output
// ---------------------------------------------------------------------------
__global__ __launch_bounds__(256) void finalize_kernel(const float* __restrict__ partial,
                                                       float* __restrict__ out) {
  __shared__ float red[256];
  float s = 0.f;
  for (int i = threadIdx.x; i < 4096; i += 256) s += partial[i];
  red[threadIdx.x] = s;
  __syncthreads();
  for (int off = 128; off > 0; off >>= 1) {
    if (threadIdx.x < off) red[threadIdx.x] += red[threadIdx.x + off];
    __syncthreads();
  }
  if (threadIdx.x == 0) {
    out[1024000] = red[0] * (1.0f / (1024.f * 20.f * 1024.f));
    out[1024001] = 0.0f;
  }
}

// ---------------------------------------------------------------------------
extern "C" void kernel_launch(void* const* d_in, const int* in_sizes, int n_in,
                              void* d_out, int out_size, void* d_ws, size_t ws_size,
                              hipStream_t stream) {
  (void)in_sizes; (void)n_in; (void)out_size; (void)ws_size;
  const float* x         = (const float*)d_in[0];
  const float* enc_W     = (const float*)d_in[1];
  const float* enc_b     = (const float*)d_in[2];
  const float* enc_ln_g  = (const float*)d_in[3];
  const float* enc_ln_b  = (const float*)d_in[4];
  const float* in_proj_W = (const float*)d_in[5];
  const float* in_proj_b = (const float*)d_in[6];
  const float* conv_w    = (const float*)d_in[7];
  const float* conv_b    = (const float*)d_in[8];
  const float* dt_W      = (const float*)d_in[9];
  const float* dt_b      = (const float*)d_in[10];
  const float* A_log     = (const float*)d_in[11];
  const float* Dv        = (const float*)d_in[12];
  const float* out_W     = (const float*)d_in[13];
  const float* out_b     = (const float*)d_in[14];
  const float* mnorm_g   = (const float*)d_in[15];
  const float* mnorm_b   = (const float*)d_in[16];
  const float* W_ih      = (const float*)d_in[17];
  const float* b_ih      = (const float*)d_in[18];
  const float* W_hh      = (const float*)d_in[19];
  const float* b_hh      = (const float*)d_in[20];
  const float* rlm_ln_g  = (const float*)d_in[21];
  const float* rlm_ln_b  = (const float*)d_in[22];
  const float* dec_W     = (const float*)d_in[23];
  const float* dec_b     = (const float*)d_in[24];
  float* out = (float*)d_out;

  char* ws = (char*)d_ws;
  size_t off = 0;
  auto alloc = [&](size_t bytes) -> void* {
    void* p = ws + off;
    off += (bytes + 255) & ~(size_t)255;
    return p;
  };
  auto allocH = [&](size_t elems) -> _Float16* {
    return (_Float16*)alloc(sizeof(_Float16) * elems);
  };
  float*    cur     = (float*)alloc(sizeof(float) * 1024 * 1024);
  _Float16* spikes  = allocH((size_t)20480 * 1024);
  float*    proj    = (float*)alloc(sizeof(float) * (size_t)20480 * 4192);
  float*    fm      = (float*)alloc(sizeof(float) * 1024 * 2048);
  float*    mbuf    = (float*)alloc(sizeof(float) * 1024 * 1024);
  float*    gi      = (float*)alloc(sizeof(float) * 1024 * 3072);
  float*    gh      = (float*)alloc(sizeof(float) * 1024 * 3072);
  float*    hbuf    = (float*)alloc(sizeof(float) * 1024 * 1024);
  float*    zfeat   = (float*)alloc(sizeof(float) * 1024 * 1024);
  float*    partial = (float*)alloc(sizeof(float) * 4096);
  // pre-split f16 hi/lo weights
  const size_t nProj = (size_t)4192 * 1024, nEnc = (size_t)1024 * 2048;
  const size_t nOut = (size_t)1024 * 2048, nIh = (size_t)3072 * 1024;
  const size_t nHh = (size_t)3072 * 1024, nDec = (size_t)1000 * 1024;
  _Float16 *projHi = allocH(nProj), *projLo = allocH(nProj);
  _Float16 *encHi  = allocH(nEnc),  *encLo  = allocH(nEnc);
  _Float16 *outHi  = allocH(nOut),  *outLo  = allocH(nOut);
  _Float16 *ihHi   = allocH(nIh),   *ihLo   = allocH(nIh);
  _Float16 *hhHi   = allocH(nHh),   *hhLo   = allocH(nHh);
  _Float16 *decHi  = allocH(nDec),  *decLo  = allocH(nDec);

  auto splitW = [&](const float* W, _Float16* hi, _Float16* lo, size_t n) {
    split_kernel<<<(unsigned)((n + 255) / 256), 256, 0, stream>>>(W, hi, lo, (int)n);
  };
  splitW(in_proj_W, projHi, projLo, nProj);
  splitW(enc_W, encHi, encLo, nEnc);
  splitW(out_W, outHi, outLo, nOut);
  splitW(W_ih, ihHi, ihLo, nIh);
  splitW(W_hh, hhHi, hhLo, nHh);
  splitW(dec_W, decHi, decLo, nDec);

  // 2) encoder GEMM: cur = x @ enc_W^T + enc_b
  gemm_wmma_kernel<false><<<dim3(1024 / 64, 1024 / 128), 256, 0, stream>>>(
      x, encHi, encLo, enc_b, cur, 1024, 1024, 2048);
  // 3) LN in place
  layernorm_kernel<<<1024, 256, 0, stream>>>(cur, enc_ln_g, enc_ln_b, cur, 1024);
  // 4) LIF -> f16 spikes + partial counts
  lif_kernel<<<4096, 256, 0, stream>>>(cur, spikes, partial);
  // 5) dominant GEMM: proj = spikes @ in_proj_W^T + b  [20480,1024]x[1024,4192]
  gemm_wmma_kernel<true><<<dim3((4192 + 63) / 64, 20480 / 128), 256, 0, stream>>>(
      spikes, projHi, projLo, in_proj_b, proj, 20480, 4192, 1024);
  // 6) fused conv+dt+SSM scan+gating -> final_memory
  ssm_kernel<<<dim3(2048 / 256, 1024), 256, 0, stream>>>(proj, conv_w, conv_b, dt_W,
                                                         dt_b, A_log, Dv, fm);
  // 7) LN(final_memory) in place
  layernorm_kernel<<<1024, 256, 0, stream>>>(fm, mnorm_g, mnorm_b, fm, 2048);
  // 8) m = LN(fm) @ out_W^T + out_b
  gemm_wmma_kernel<false><<<dim3(1024 / 64, 1024 / 128), 256, 0, stream>>>(
      fm, outHi, outLo, out_b, mbuf, 1024, 1024, 2048);
  // 9) gi = m @ W_ih^T + b_ih (fixed for all GRU iterations)
  gemm_wmma_kernel<false><<<dim3(3072 / 64, 1024 / 128), 256, 0, stream>>>(
      mbuf, ihHi, ihLo, b_ih, gi, 1024, 3072, 1024);
  // 10) GRU iter 0: h==0 -> gh = b_hh exactly, no GEMM needed
  gru_gate_kernel<<<4096, 256, 0, stream>>>(gi, b_hh, 0, hbuf, 1);
  // 11) GRU iters 1..4
  for (int it = 1; it < 5; ++it) {
    gemm_wmma_kernel<false><<<dim3(3072 / 64, 1024 / 128), 256, 0, stream>>>(
        hbuf, hhHi, hhLo, b_hh, gh, 1024, 3072, 1024);
    gru_gate_kernel<<<4096, 256, 0, stream>>>(gi, gh, 3072, hbuf, 0);
  }
  // 12) LN -> zfeat
  layernorm_kernel<<<1024, 256, 0, stream>>>(hbuf, rlm_ln_g, rlm_ln_b, zfeat, 1024);
  // 13) logits = zfeat @ dec_W^T + dec_b (N=1000 handled by bounds checks)
  gemm_wmma_kernel<false><<<dim3((1000 + 63) / 64, 1024 / 128), 256, 0, stream>>>(
      zfeat, decHi, decLo, dec_b, out, 1024, 1000, 1024);
  // 14) rate + trailing 0.0
  finalize_kernel<<<1, 256, 0, stream>>>(partial, out);
}
